// MultiHeadSelfAttention_22668837388979
// MI455X (gfx1250) — compile-verified, matrix operand order checked by reading, not run on hardware
//
#include <hip/hip_runtime.h>
#include <hip/hip_bf16.h>
#include <math.h>

// MHSA forward for MI455X (gfx1250, wave32, WMMA bf16 -> f32 accumulate).
//
// Shapes: B=2, S=2048, E=1024, H=16, D=64.
// Pipeline:
//   1) cvt: fp32 -> bf16 for X, Wqkv, Wo
//   2) qkv_gemm: 128x64 macro-tile / block, B-tile staged in LDS with
//      double-buffered ASYNC-to-LDS copies (ASYNCcnt), 8 waves x 16-row tiles
//   3) flash_attn: per-wave 16-query block, online softmax over 32-key blocks,
//      P re-fragmented through LDS, ghost denominator
//   4) out_gemm: same macro-tile structure as qkv_gemm -> f32 d_out
//
// Scheduling: all fragments of a chunk are batched into registers before the
// WMMA burst so ds/global loads issue as clauses with progressive waits.

typedef __attribute__((ext_vector_type(16))) __bf16 v16bf;
typedef __attribute__((ext_vector_type(8)))  float  v8f;
typedef int vi4 __attribute__((vector_size(16)));   // matches async-builtin param

#define WMMA_BF16(A_, B_, C_) \
  __builtin_amdgcn_wmma_f32_16x16x32_bf16(false, (A_), false, (B_), (short)0, (C_), false, false)

#define BATCH 2
#define SEQ   2048
#define EMBED 1024
#define NHEAD 16
#define HDIM  64
#define KC    64            // k-chunk staged in LDS per GEMM iteration
#define BPAD  16            // row pad (elements): stride 80*2B=160B, 32B-aligned
#define BSTRIDE (KC + BPAD)

// ---- CDNA5 async global->LDS copy (ASYNCcnt) with portable fallback --------
#if __has_builtin(__builtin_amdgcn_global_load_async_to_lds_b128)
#define HAVE_ASYNC_LDS 1
#define ASYNC_CP16(SRC, DST)                                                   \
  __builtin_amdgcn_global_load_async_to_lds_b128(                              \
      (__attribute__((address_space(1))) vi4*)(SRC),                           \
      (__attribute__((address_space(3))) vi4*)(DST), 0, 0)
#else
#define HAVE_ASYNC_LDS 0
struct __align__(16) u128_t { unsigned int x[4]; };
#define ASYNC_CP16(SRC, DST) (*(u128_t*)(DST) = *(const u128_t*)(SRC))
#endif

#if HAVE_ASYNC_LDS && __has_builtin(__builtin_amdgcn_s_wait_asynccnt)
#define WAIT_ASYNC(N) __builtin_amdgcn_s_wait_asynccnt(N)
#else
#define WAIT_ASYNC(N) ((void)0)
#endif

// ---------------------------------------------------------------- convert
__global__ __launch_bounds__(256) void cvt_f32_bf16(const float* __restrict__ src,
                                                    __bf16* __restrict__ dst, int n) {
  int i = blockIdx.x * 256 + threadIdx.x;
  if (i < n) dst[i] = (__bf16)src[i];
}

// ---------------------------------------------------------------- QKV GEMM
// Block: 256 threads (8 waves) -> macro tile M=128, N=64 (one head's dims).
// B tile (64 x KC) double-buffered in LDS via async copies; each wave computes
// a 16x64 strip with 4 f32 accumulators, A fragments loaded directly (rows are
// private to the wave; X stays resident in L2).
__global__ __launch_bounds__(256) void qkv_gemm(
    const __bf16* __restrict__ X,      // [4096,1024]
    const __bf16* __restrict__ W,      // [3072,1024]
    const float*  __restrict__ bias,   // [3072]
    __bf16* __restrict__ Qb,           // [B,H,S,64]
    __bf16* __restrict__ Kb,           // [B,H,S,64]
    __bf16* __restrict__ Vt)           // [B,H,64,S]
{
  __shared__ __align__(128) __bf16 Bt[2][64][BSTRIDE];   // 2 x 10KB

  const int tid  = threadIdx.x;
  const int lane = tid & 31;
  const int wv   = tid >> 5;                 // wave 0..7
  const int m_base = blockIdx.x * 128 + wv * 16;
  const int n_base = blockIdx.y * 64;
  const int lm = lane & 15;
  const int hi = lane >> 4;

  // B-tile copy: 512 16-byte chunks (64 rows x 8 chunks/row), 2 per thread
  const int ci0 = tid;                        // chunk ids: tid, tid+256
  const __bf16* Wbase = W + n_base * EMBED;

  const __bf16* arow = X + (m_base + lm) * EMBED + hi * 16;

  v8f acc[4] = {};

  // prologue: stage k-chunk 0 into buffer 0
  {
    const int r0 = ci0 >> 3, c0 = (ci0 & 7) * 8;
    const int r1 = (ci0 + 256) >> 3, c1 = ((ci0 + 256) & 7) * 8;
    ASYNC_CP16(Wbase + r0 * EMBED + c0, &Bt[0][r0][c0]);
    ASYNC_CP16(Wbase + r1 * EMBED + c1, &Bt[0][r1][c1]);
  }

  const int NKC = EMBED / KC;                 // 16 chunks
  for (int i = 0; i < NKC; ++i) {
    const int cur = i & 1;
    if (i + 1 < NKC) {                        // stage next chunk into other buffer
      const int k0 = (i + 1) * KC;
      const int r0 = ci0 >> 3, c0 = (ci0 & 7) * 8;
      const int r1 = (ci0 + 256) >> 3, c1 = ((ci0 + 256) & 7) * 8;
      ASYNC_CP16(Wbase + r0 * EMBED + k0 + c0, &Bt[cur ^ 1][r0][c0]);
      ASYNC_CP16(Wbase + r1 * EMBED + k0 + c1, &Bt[cur ^ 1][r1][c1]);
      WAIT_ASYNC(2);                          // chunk i's copies retired
    } else {
      WAIT_ASYNC(0);
    }
    __syncthreads();                          // buffer cur valid for all waves

    // batch all fragments, then burst the 8 WMMAs
    v16bf a0 = *(const v16bf*)(arow + i * KC);
    v16bf a1 = *(const v16bf*)(arow + i * KC + 32);
    v16bf bf[8];
#pragma unroll
    for (int t = 0; t < 4; ++t) {
      bf[2 * t]     = *(const v16bf*)&Bt[cur][t * 16 + lm][hi * 16];
      bf[2 * t + 1] = *(const v16bf*)&Bt[cur][t * 16 + lm][32 + hi * 16];
    }
#pragma unroll
    for (int t = 0; t < 4; ++t) {
      acc[t] = WMMA_BF16(a0, bf[2 * t],     acc[t]);
      acc[t] = WMMA_BF16(a1, bf[2 * t + 1], acc[t]);
    }
    __syncthreads();                          // readers done before overwrite
  }

  // bias + scatter to Q / K / V^T
  const int c = n_base >> 10;                 // 0=Q 1=K 2=V
  const int h = (n_base & 1023) >> 6;
#pragma unroll
  for (int t = 0; t < 4; ++t) {
    const int   d  = t * 16 + lm;
    const float bv = bias[n_base + t * 16 + lm];
#pragma unroll
    for (int r = 0; r < 8; ++r) {
      const int m  = m_base + r + 8 * hi;
      const int bb = m >> 11;
      const int s  = m & (SEQ - 1);
      const __bf16 o = (__bf16)(acc[t][r] + bv);
      if (c == 0)      Qb[(((bb * NHEAD + h) * SEQ) + s) * HDIM + d] = o;
      else if (c == 1) Kb[(((bb * NHEAD + h) * SEQ) + s) * HDIM + d] = o;
      else             Vt[(((bb * NHEAD + h) * HDIM) + d) * SEQ + s] = o;
    }
  }
}

// ---------------------------------------------------------------- attention
// One wave per (b, h, 16-query block). Key loop in blocks of 32.
__global__ __launch_bounds__(128) void flash_attn(
    const __bf16* __restrict__ Qb,     // [B,H,S,64]
    const __bf16* __restrict__ Kb,     // [B,H,S,64]
    const __bf16* __restrict__ Vt,     // [B,H,64,S]
    const float*  __restrict__ ghost,  // [H]
    __bf16* __restrict__ Abuf)         // [B,S,H*64]
{
  __shared__ __align__(64) __bf16 Plds[4][16][32];

  const int lane  = threadIdx.x & 31;
  const int wslot = threadIdx.x >> 5;
  const int wid   = blockIdx.x * 4 + wslot;
  const int b  = wid >> 11;
  const int h  = (wid >> 7) & 15;
  const int qb = wid & 127;
  const int lm = lane & 15;
  const int hi = lane >> 4;

  const __bf16* Qrow  = Qb + (((b * NHEAD + h) * SEQ) + qb * 16 + lm) * HDIM + hi * 16;
  const __bf16* Kbase = Kb + ((b * NHEAD + h) * SEQ) * HDIM + lm * HDIM + hi * 16;
  const __bf16* Vbase = Vt + ((b * NHEAD + h) * HDIM + lm) * SEQ + hi * 16;

  // Q fragments are loop-invariant: load once.
  v16bf qa0 = *(const v16bf*)(Qrow);
  v16bf qa1 = *(const v16bf*)(Qrow + 32);

  float m_r[8], l_r[8];
  v8f O[4] = {};
#pragma unroll
  for (int r = 0; r < 8; ++r) { m_r[r] = -3.0e38f; l_r[r] = 0.0f; }

  const float scale = 0.125f;               // 1/sqrt(64)
  const int nkb = (qb >> 1) + 1;            // causal coverage

  for (int kb = 0; kb < nkb; ++kb) {
    // batch 4 K fragments, then 4 score WMMAs
    v16bf kf[4];
    kf[0] = *(const v16bf*)(Kbase + (kb * 32 + 0)  * HDIM);
    kf[1] = *(const v16bf*)(Kbase + (kb * 32 + 0)  * HDIM + 32);
    kf[2] = *(const v16bf*)(Kbase + (kb * 32 + 16) * HDIM);
    kf[3] = *(const v16bf*)(Kbase + (kb * 32 + 16) * HDIM + 32);

    v8f s0 = {}, s1 = {};
    s0 = WMMA_BF16(qa0, kf[0], s0);
    s0 = WMMA_BF16(qa1, kf[1], s0);
    s1 = WMMA_BF16(qa0, kf[2], s1);
    s1 = WMMA_BF16(qa1, kf[3], s1);

    // prefetch V fragments for this key block while doing softmax VALU work
    v16bf vf[4];
#pragma unroll
    for (int t = 0; t < 4; ++t)
      vf[t] = *(const v16bf*)(Vbase + t * 16 * SEQ + kb * 32);

    const int key0 = kb * 32 + lm;
    const int key1 = key0 + 16;
#pragma unroll
    for (int r = 0; r < 8; ++r) {
      const int row = qb * 16 + r + 8 * hi;
      float x0 = (key0 <= row) ? s0[r] * scale : -3.0e38f;
      float x1 = (key1 <= row) ? s1[r] * scale : -3.0e38f;

      float mx = fmaxf(x0, x1);
      mx = fmaxf(mx, __shfl_xor(mx, 1, 32));
      mx = fmaxf(mx, __shfl_xor(mx, 2, 32));
      mx = fmaxf(mx, __shfl_xor(mx, 4, 32));
      mx = fmaxf(mx, __shfl_xor(mx, 8, 32));

      const float mn    = fmaxf(m_r[r], mx);
      const float alpha = __expf(m_r[r] - mn);
      const float p0 = __expf(x0 - mn);
      const float p1 = __expf(x1 - mn);
      float rs = p0 + p1;
      rs += __shfl_xor(rs, 1, 32);
      rs += __shfl_xor(rs, 2, 32);
      rs += __shfl_xor(rs, 4, 32);
      rs += __shfl_xor(rs, 8, 32);

      l_r[r] = l_r[r] * alpha + rs;
      m_r[r] = mn;
#pragma unroll
      for (int t = 0; t < 4; ++t) O[t][r] = O[t][r] * alpha;

      Plds[wslot][r + 8 * hi][lm]      = (__bf16)p0;
      Plds[wslot][r + 8 * hi][16 + lm] = (__bf16)p1;
    }

    v16bf pa = *(const v16bf*)&Plds[wslot][lm][hi * 16];
#pragma unroll
    for (int t = 0; t < 4; ++t) O[t] = WMMA_BF16(pa, vf[t], O[t]);
  }

  const float gh = ghost[h];
#pragma unroll
  for (int r = 0; r < 8; ++r) {
    const float inv  = 1.0f / (l_r[r] + gh);
    const int   srow = qb * 16 + r + 8 * hi;
#pragma unroll
    for (int t = 0; t < 4; ++t) {
      Abuf[(b * SEQ + srow) * EMBED + h * HDIM + t * 16 + lm] = (__bf16)(O[t][r] * inv);
    }
  }
}

// ---------------------------------------------------------------- out GEMM
// Out = A @ Wo^T + bias; same LDS macro-tile structure as qkv_gemm.
__global__ __launch_bounds__(256) void out_gemm(
    const __bf16* __restrict__ A,      // [4096,1024]
    const __bf16* __restrict__ W,      // [1024,1024]
    const float*  __restrict__ bias,   // [1024]
    float* __restrict__ out)           // [4096,1024]
{
  __shared__ __align__(128) __bf16 Bt[2][64][BSTRIDE];

  const int tid  = threadIdx.x;
  const int lane = tid & 31;
  const int wv   = tid >> 5;
  const int m_base = blockIdx.x * 128 + wv * 16;
  const int n_base = blockIdx.y * 64;
  const int lm = lane & 15;
  const int hi = lane >> 4;

  const int ci0 = tid;
  const __bf16* Wbase = W + n_base * EMBED;
  const __bf16* arow  = A + (m_base + lm) * EMBED + hi * 16;

  v8f acc[4] = {};

  {
    const int r0 = ci0 >> 3, c0 = (ci0 & 7) * 8;
    const int r1 = (ci0 + 256) >> 3, c1 = ((ci0 + 256) & 7) * 8;
    ASYNC_CP16(Wbase + r0 * EMBED + c0, &Bt[0][r0][c0]);
    ASYNC_CP16(Wbase + r1 * EMBED + c1, &Bt[0][r1][c1]);
  }

  const int NKC = EMBED / KC;
  for (int i = 0; i < NKC; ++i) {
    const int cur = i & 1;
    if (i + 1 < NKC) {
      const int k0 = (i + 1) * KC;
      const int r0 = ci0 >> 3, c0 = (ci0 & 7) * 8;
      const int r1 = (ci0 + 256) >> 3, c1 = ((ci0 + 256) & 7) * 8;
      ASYNC_CP16(Wbase + r0 * EMBED + k0 + c0, &Bt[cur ^ 1][r0][c0]);
      ASYNC_CP16(Wbase + r1 * EMBED + k0 + c1, &Bt[cur ^ 1][r1][c1]);
      WAIT_ASYNC(2);
    } else {
      WAIT_ASYNC(0);
    }
    __syncthreads();

    v16bf a0 = *(const v16bf*)(arow + i * KC);
    v16bf a1 = *(const v16bf*)(arow + i * KC + 32);
    v16bf bf[8];
#pragma unroll
    for (int t = 0; t < 4; ++t) {
      bf[2 * t]     = *(const v16bf*)&Bt[cur][t * 16 + lm][hi * 16];
      bf[2 * t + 1] = *(const v16bf*)&Bt[cur][t * 16 + lm][32 + hi * 16];
    }
#pragma unroll
    for (int t = 0; t < 4; ++t) {
      acc[t] = WMMA_BF16(a0, bf[2 * t],     acc[t]);
      acc[t] = WMMA_BF16(a1, bf[2 * t + 1], acc[t]);
    }
    __syncthreads();
  }

#pragma unroll
  for (int t = 0; t < 4; ++t) {
    const float bv = bias[n_base + t * 16 + lm];
#pragma unroll
    for (int r = 0; r < 8; ++r) {
      out[(m_base + r + 8 * hi) * EMBED + n_base + t * 16 + lm] = acc[t][r] + bv;
    }
  }
}

// ---------------------------------------------------------------- launch
extern "C" void kernel_launch(void* const* d_in, const int* in_sizes, int n_in,
                              void* d_out, int out_size, void* d_ws, size_t ws_size,
                              hipStream_t stream) {
  const float* X     = (const float*)d_in[0];  // [2,2048,1024]
  const float* Wqkv  = (const float*)d_in[1];  // [3072,1024]
  const float* Bqkv  = (const float*)d_in[2];  // [3072]
  const float* Wo    = (const float*)d_in[3];  // [1024,1024]
  const float* Bo    = (const float*)d_in[4];  // [1024]
  const float* ghost = (const float*)d_in[5];  // [16]

  char* ws = (char*)d_ws;
  __bf16* Xbf = (__bf16*)(ws + 0);             // 4096*1024*2  =  8 MiB
  __bf16* Wqb = (__bf16*)(ws + 8388608);       // 3072*1024*2  =  6 MiB
  __bf16* Wob = (__bf16*)(ws + 14680064);      // 1024*1024*2  =  2 MiB
  __bf16* Qb  = (__bf16*)(ws + 16777216);      //                 8 MiB
  __bf16* Kb  = (__bf16*)(ws + 25165824);      //                 8 MiB
  __bf16* Vt  = (__bf16*)(ws + 33554432);      //                 8 MiB
  __bf16* Ab  = (__bf16*)(ws + 41943040);      //                 8 MiB

  cvt_f32_bf16<<<(4194304 + 255) / 256, 256, 0, stream>>>(X,    Xbf, 4194304);
  cvt_f32_bf16<<<(3145728 + 255) / 256, 256, 0, stream>>>(Wqkv, Wqb, 3145728);
  cvt_f32_bf16<<<(1048576 + 255) / 256, 256, 0, stream>>>(Wo,   Wob, 1048576);

  // QKV: grid (M/128, N/64) = (32, 48)
  qkv_gemm<<<dim3(32, 48), 256, 0, stream>>>(Xbf, Wqb, Bqkv, Qb, Kb, Vt);

  // attention: 2*16*128 = 4096 waves, 4 waves/block
  flash_attn<<<4096 / 4, 128, 0, stream>>>(Qb, Kb, Vt, ghost, Ab);

  // out: grid (M/128, N/64) = (32, 16)
  out_gemm<<<dim3(32, 16), 256, 0, stream>>>(Ab, Wob, Bo, (float*)d_out);
}
